// ModernBertAttention_53455162966700
// MI455X (gfx1250) — matrix operand order repaired and checked by reading, not compiled, not run on hardware
//
#include <hip/hip_runtime.h>
#include <hip/hip_fp16.h>

typedef __attribute__((ext_vector_type(16))) _Float16 v16h;
typedef __attribute__((ext_vector_type(8)))  float    v8f;
typedef __attribute__((ext_vector_type(2)))  __fp16   v2fp16;

#define DEVINL __device__ __forceinline__

namespace {

constexpr int Bc  = 4;
constexpr int Sc  = 2048;
constexpr int Dc  = 768;
constexpr int Hc  = 12;
constexpr int Dhc = 64;
constexpr int Nqkv = 3 * Dc;           // 2304
constexpr float kNegBig = -1.0e9f;

// ---- WMMA helpers ------------------------------------------------------
// B-fragment (32x16 f16): lane holds one column; halfs K = g*8..g*8+7 and
// 16+g*8..+7 (g = l/16). Caller passes p = colBase + g*8 -> chunks at p, p+16.
DEVINL v16h ldsFragB(const _Float16* p) {
  union { v16h h; uint4 u[2]; } f;
  f.u[0] = *reinterpret_cast<const uint4*>(p);
  f.u[1] = *reinterpret_cast<const uint4*>(p + 16);
  return f.h;
}
// A-fragment (16x32 f16): lane l holds row l%16; halfs K = g*8..g*8+7 and
// 16+g*8..+7 (g = l/16). Caller passes p = rowBase + g*8 -> chunks at p, p+16.
DEVINL v16h ldsFragA(const _Float16* p) {
  union { v16h h; uint4 u[2]; } f;
  f.u[0] = *reinterpret_cast<const uint4*>(p);
  f.u[1] = *reinterpret_cast<const uint4*>(p + 16);
  return f.h;
}
DEVINL v8f wmma16(v16h a, v16h b, v8f c) {
  return __builtin_amdgcn_wmma_f32_16x16x32_f16(false, a, false, b,
                                                (short)0, c, false, false);
}
DEVINL v8f zero8() { v8f z = {0.f,0.f,0.f,0.f,0.f,0.f,0.f,0.f}; return z; }

// packed f32 pair -> f16 pair (v_cvt_pk_rtz_f16_f32), returned as u32 bits
DEVINL uint32_t pkHalf2(float lo, float hi) {
  union { v2fp16 h; uint32_t u; } cv;
  cv.h = __builtin_amdgcn_cvt_pkrtz(lo, hi);
  return cv.u;
}

// ---- CDNA5 async global -> LDS (ASYNCcnt path) -------------------------
// LDS dest = byte offset ldsOff (VGPR); global src = sbase (SGPR) + voff (VGPR)
DEVINL void asyncLoadB128(uint32_t ldsOff, const void* sbase, uint32_t voff) {
  asm volatile("global_load_async_to_lds_b128 %0, %1, %2"
               :: "v"(ldsOff), "v"(voff), "s"(sbase)
               : "memory");
}
DEVINL void waitAsync0() {
  asm volatile("s_wait_asynccnt 0" ::: "memory");
}
// LDS byte offset of a generic pointer to __shared__ (low 32 bits of the
// flat address are the wave-relative LDS offset on gfx1250).
DEVINL uint32_t ldsOffsetOf(const void* p) {
  return (uint32_t)(uintptr_t)p;
}

// ---- Kernel 1: QKV GEMM (f16 WMMA, f32 acc) + fused RoPE ---------------
__global__ __launch_bounds__(256)
void qkv_rope_kernel(const float* __restrict__ x, const float* __restrict__ wqkv,
                     _Float16* __restrict__ Qw, _Float16* __restrict__ Kw,
                     _Float16* __restrict__ Vw) {
  constexpr int LDA = 40;                       // halfs per LDS row (32 + pad)
  __shared__ __align__(16) _Float16 As[128 * LDA];
  __shared__ __align__(16) _Float16 Bs[128 * LDA];

  const int tid  = threadIdx.x;
  const int lane = tid & 31;
  const int wave = tid >> 5;
  const int l16  = lane & 15;
  const int g    = lane >> 4;
  const int wr   = wave >> 1;                   // 0..3 row-wave
  const int wc   = wave & 1;                    // 0..1 col-wave
  const int mb   = blockIdx.x * 128;
  const int nb   = blockIdx.y * 128;

  v8f acc[2][4];
  #pragma unroll
  for (int rt = 0; rt < 2; ++rt)
    #pragma unroll
    for (int ct = 0; ct < 4; ++ct) acc[rt][ct] = zero8();

  uint32_t* dA = (uint32_t*)As;
  uint32_t* dB = (uint32_t*)Bs;

  for (int kk = 0; kk < Dc; kk += 32) {
    // stage A: x[mb+r][kk+c2..c2+1] -> As[r][c2..c2+1]  (packed f32->f16)
    #pragma unroll
    for (int i = 0; i < 8; ++i) {
      int idx = i * 256 + tid;                  // 2048 half-pairs
      int r = idx >> 4, cp = idx & 15;          // 16 pairs per row
      const float2 f2 = *reinterpret_cast<const float2*>(
          &x[(size_t)(mb + r) * Dc + kk + cp * 2]);
      dA[r * (LDA / 2) + cp] = pkHalf2(f2.x, f2.y);
    }
    // stage B transposed: wqkv[kk+k2..k2+1][nb+n] -> Bs[n][k2..k2+1]
    #pragma unroll
    for (int i = 0; i < 8; ++i) {
      int idx = i * 256 + tid;                  // 16 k-pairs x 128 n
      int kp = idx >> 7, n = idx & 127;
      float f0 = wqkv[(size_t)(kk + kp * 2)     * Nqkv + nb + n];
      float f1 = wqkv[(size_t)(kk + kp * 2 + 1) * Nqkv + nb + n];
      dB[n * (LDA / 2) + kp] = pkHalf2(f0, f1);
    }
    if (kk + 32 < Dc) {   // hint next tiles toward L2 (global_prefetch_b8)
      __builtin_prefetch(&x[(size_t)(mb + (tid >> 1)) * Dc + kk + 32], 0, 1);
      __builtin_prefetch(&wqkv[(size_t)(kk + 32 + (tid >> 7)) * Nqkv + nb + (tid & 127)], 0, 1);
    }
    __syncthreads();

    v16h a0 = ldsFragA(&As[(wr * 32 +  0 + l16) * LDA + g * 8]);
    v16h a1 = ldsFragA(&As[(wr * 32 + 16 + l16) * LDA + g * 8]);
    #pragma unroll
    for (int ct = 0; ct < 4; ++ct) {
      v16h bf = ldsFragB(&Bs[(wc * 64 + ct * 16 + l16) * LDA + g * 8]);
      acc[0][ct] = wmma16(a0, bf, acc[0][ct]);
      acc[1][ct] = wmma16(a1, bf, acc[1][ct]);
    }
    __syncthreads();
  }

  // epilogue: this wave's 64 columns are exactly one head of one section
  const int n64 = nb + wc * 64;
  const int g64 = n64 >> 6;
  const int sec = g64 / Hc;                     // 0=q 1=k 2=v
  const int h   = g64 % Hc;

  if (sec == 2) {
    #pragma unroll
    for (int rt = 0; rt < 2; ++rt)
      #pragma unroll
      for (int ct = 0; ct < 4; ++ct)
        #pragma unroll
        for (int r = 0; r < 8; ++r) {
          int m  = mb + wr * 32 + rt * 16 + r + 8 * g;
          int bb = m >> 11, s = m & (Sc - 1);
          int d  = ct * 16 + l16;
          Vw[((size_t)(bb * Hc + h) * Sc + s) * Dhc + d] = (_Float16)acc[rt][ct][r];
        }
  } else {
    _Float16* dst = (sec == 0) ? Qw : Kw;
    const float mul = (sec == 0) ? 0.125f : 1.0f;   // fold 1/sqrt(Dh) into Q
    #pragma unroll
    for (int ct = 0; ct < 2; ++ct) {
      int   j    = ct * 16 + l16;                    // rope index in [0,32)
      float freq = __expf(-0.28782313271f * (float)j); // 10000^(-j/32)
      #pragma unroll
      for (int rt = 0; rt < 2; ++rt)
        #pragma unroll
        for (int r = 0; r < 8; ++r) {
          int m  = mb + wr * 32 + rt * 16 + r + 8 * g;
          int bb = m >> 11, s = m & (Sc - 1);
          float sn, cs;
          __sincosf((float)s * freq, &sn, &cs);
          float x1 = acc[rt][ct][r];
          float x2 = acc[rt][ct + 2][r];               // pair d <-> d+32
          size_t base = ((size_t)(bb * Hc + h) * Sc + s) * Dhc;
          dst[base + j]      = (_Float16)((x1 * cs - x2 * sn) * mul);
          dst[base + j + 32] = (_Float16)((x2 * cs + x1 * sn) * mul);
        }
    }
  }
}

// ---- Kernel 2: sliding-window attention --------------------------------
__global__ __launch_bounds__(256)
void swa_kernel(const _Float16* __restrict__ Qw, const _Float16* __restrict__ Kw,
                const _Float16* __restrict__ Vw, const int* __restrict__ pmask,
                _Float16* __restrict__ Cw) {
  constexpr int LQK = 72;    // halfs per Q/K LDS row (64 + 8 pad)
  constexpr int LPV = 136;   // halfs per P/Vt LDS row (128 + 8 pad)
  __shared__ __align__(16) char smem[56320];
  _Float16* Qs = (_Float16*)(smem);              // 128 x 72  (phase 1)
  _Float16* Ks = (_Float16*)(smem + 18432);      // 256 x 72  (phase 1)
  float*    Mf = (float*)   (smem + 55296);      // 256 mask floats
  _Float16* Pb = (_Float16*)(smem);              // 128 x 136 (phase 2)
  _Float16* Vt = (_Float16*)(smem + 34816);      // 64 x 136  (phase 2)

  const int tid  = threadIdx.x;
  const int lane = tid & 31;
  const int wave = tid >> 5;
  const int l16  = lane & 15;
  const int g    = lane >> 4;
  const int qb   = blockIdx.x * 128;
  const int h    = blockIdx.y;
  const int b    = blockIdx.z;
  const int kb   = qb - 64;                      // first key of the 256 span
  const size_t headBase = (size_t)(b * Hc + h) * Sc * Dhc;

  // stage Q tile via CDNA5 async global->LDS (always in bounds)
  {
    const _Float16* gq = Qw + headBase + (size_t)qb * Dhc;
    #pragma unroll
    for (int i = 0; i < 4; ++i) {
      int idx = i * 256 + tid;                   // 1024 x 16B chunks
      int r = idx >> 3, c = idx & 7;             // 8 chunks per 128B row
      asyncLoadB128(ldsOffsetOf(Qs + r * LQK) + c * 16, gq,
                    (uint32_t)(r * 128 + c * 16));
    }
  }
  // stage K span via async loads; clamp OOB rows (they are masked to -1e9,
  // clamping just guarantees finite data)
  {
    const _Float16* gk = Kw + headBase;
    #pragma unroll
    for (int i = 0; i < 8; ++i) {
      int idx = i * 256 + tid;                   // 2048 x 16B chunks
      int r = idx >> 3, c = idx & 7;
      int j = kb + r;
      j = (j < 0) ? 0 : ((j >= Sc) ? Sc - 1 : j);
      asyncLoadB128(ldsOffsetOf(Ks + r * LQK) + c * 16, gk,
                    (uint32_t)(j * 128 + c * 16));
    }
  }
  { int j = kb + tid; Mf[tid] = (j >= 0 && j < Sc) ? (float)pmask[b * Sc + j] : 0.0f; }
  waitAsync0();
  __syncthreads();

  // ---- scores: each wave computes 16 query rows x 256 keys -------------
  v16h aQ0 = ldsFragA(&Qs[(wave * 16 + l16) * LQK + g * 8]);
  v16h aQ1 = ldsFragA(&Qs[(wave * 16 + l16) * LQK + 32 + g * 8]);
  v8f sc[16];
  #pragma unroll
  for (int t = 0; t < 16; ++t) {
    v16h b0 = ldsFragB(&Ks[(t * 16 + l16) * LQK + g * 8]);
    v8f z = wmma16(aQ0, b0, zero8());
    v16h b1 = ldsFragB(&Ks[(t * 16 + l16) * LQK + 32 + g * 8]);
    sc[t] = wmma16(aQ1, b1, z);
  }

  // ---- mask + exact softmax (row = (VGPR r, half-wave)) ----------------
  #pragma unroll
  for (int r = 0; r < 8; ++r) {
    int iq = qb + wave * 16 + r + 8 * g;
    float mx = kNegBig;
    #pragma unroll
    for (int t = 0; t < 16; ++t) {
      int jk = kb + t * 16 + l16;
      bool ok = (jk >= 0) && (jk < Sc) && (jk - iq <= 64) && (iq - jk <= 64) &&
                (Mf[t * 16 + l16] > 0.5f);
      float s = ok ? sc[t][r] : kNegBig;
      sc[t][r] = s;
      mx = fmaxf(mx, s);
    }
    #pragma unroll
    for (int d = 1; d < 16; d <<= 1) mx = fmaxf(mx, __shfl_xor(mx, d, 32));
    float sum = 0.0f;
    #pragma unroll
    for (int t = 0; t < 16; ++t) {
      float e = __expf(sc[t][r] - mx);
      sc[t][r] = e;
      sum += e;
    }
    #pragma unroll
    for (int d = 1; d < 16; d <<= 1) sum += __shfl_xor(sum, d, 32);
    float inv = 1.0f / sum;
    #pragma unroll
    for (int t = 0; t < 16; ++t) sc[t][r] *= inv;
  }

  // ---- P @ V over two 128-key chunks (LDS reused) ----------------------
  v8f outAcc[4];
  #pragma unroll
  for (int ct = 0; ct < 4; ++ct) outAcc[ct] = zero8();

  #pragma unroll 1
  for (int c = 0; c < 2; ++c) {
    __syncthreads();   // previous-phase LDS reads complete before overwrite
    // stage V chunk transposed: Vt[dh][key]; clamp OOB (p == 0 there)
    #pragma unroll
    for (int i = 0; i < 16; ++i) {
      int idx = i * 256 + tid;
      int key = idx >> 5, cu = idx & 31;
      int dh = cu * 2;
      int j = kb + c * 128 + key;
      j = (j < 0) ? 0 : ((j >= Sc) ? Sc - 1 : j);
      uint32_t v = *(const uint32_t*)(Vw + headBase + (size_t)j * Dhc + dh);
      union { uint32_t u; _Float16 hh[2]; } cvt; cvt.u = v;
      Vt[dh * LPV + key]       = cvt.hh[0];
      Vt[(dh + 1) * LPV + key] = cvt.hh[1];
    }
    // write this chunk's probabilities: Pb[query][key] f16
    #pragma unroll
    for (int t8 = 0; t8 < 8; ++t8)
      #pragma unroll
      for (int r = 0; r < 8; ++r) {
        int row = wave * 16 + r + 8 * g;
        Pb[row * LPV + t8 * 16 + l16] = (_Float16)sc[c * 8 + t8][r];
      }
    __syncthreads();

    #pragma unroll
    for (int tt = 0; tt < 4; ++tt) {
      v16h aP = ldsFragA(&Pb[(wave * 16 + l16) * LPV + tt * 32 + g * 8]);
      #pragma unroll
      for (int ct = 0; ct < 4; ++ct) {
        v16h bV = ldsFragB(&Vt[(ct * 16 + l16) * LPV + tt * 32 + g * 8]);
        outAcc[ct] = wmma16(aP, bV, outAcc[ct]);
      }
    }
  }

  // context out: Cw[b*S + i][h*64 + d]  (f16)
  #pragma unroll
  for (int ct = 0; ct < 4; ++ct)
    #pragma unroll
    for (int r = 0; r < 8; ++r) {
      int iq  = qb + wave * 16 + r + 8 * g;
      int col = h * Dhc + ct * 16 + l16;
      Cw[(size_t)(b * Sc + iq) * Dc + col] = (_Float16)outAcc[ct][r];
    }
}

// ---- Kernel 3: output projection GEMM ----------------------------------
__global__ __launch_bounds__(256)
void out_proj_kernel(const _Float16* __restrict__ Cw, const float* __restrict__ wout,
                     float* __restrict__ out) {
  constexpr int LDA = 40;
  __shared__ __align__(16) _Float16 As[128 * LDA];
  __shared__ __align__(16) _Float16 Bs[128 * LDA];

  const int tid  = threadIdx.x;
  const int lane = tid & 31;
  const int wave = tid >> 5;
  const int l16  = lane & 15;
  const int g    = lane >> 4;
  const int wr   = wave >> 1;
  const int wc   = wave & 1;
  const int mb   = blockIdx.x * 128;
  const int nb   = blockIdx.y * 128;

  v8f acc[2][4];
  #pragma unroll
  for (int rt = 0; rt < 2; ++rt)
    #pragma unroll
    for (int ct = 0; ct < 4; ++ct) acc[rt][ct] = zero8();

  uint32_t* dB = (uint32_t*)Bs;

  for (int kk = 0; kk < Dc; kk += 32) {
    // stage A from f16 context via async global->LDS copy
    {
      const _Float16* gA = Cw + (size_t)mb * Dc + kk;   // row stride 768 halfs
      #pragma unroll
      for (int i = 0; i < 2; ++i) {
        int idx = i * 256 + tid;         // 512 x 16B chunks (128 rows x 4)
        int r = idx >> 2, c = idx & 3;
        asyncLoadB128(ldsOffsetOf(As + r * LDA) + c * 16, gA,
                      (uint32_t)(r * (Dc * 2) + c * 16));
      }
    }
    // stage B transposed: wout[kk+k2..k2+1][nb+n] -> Bs[n][k2..k2+1]
    #pragma unroll
    for (int i = 0; i < 8; ++i) {
      int idx = i * 256 + tid;
      int kp = idx >> 7, n = idx & 127;
      float f0 = wout[(size_t)(kk + kp * 2)     * Dc + nb + n];
      float f1 = wout[(size_t)(kk + kp * 2 + 1) * Dc + nb + n];
      dB[n * (LDA / 2) + kp] = pkHalf2(f0, f1);
    }
    if (kk + 32 < Dc) {
      __builtin_prefetch(&wout[(size_t)(kk + 32 + (tid >> 7)) * Dc + nb + (tid & 127)], 0, 1);
    }
    waitAsync0();
    __syncthreads();

    v16h a0 = ldsFragA(&As[(wr * 32 +  0 + l16) * LDA + g * 8]);
    v16h a1 = ldsFragA(&As[(wr * 32 + 16 + l16) * LDA + g * 8]);
    #pragma unroll
    for (int ct = 0; ct < 4; ++ct) {
      v16h bf = ldsFragB(&Bs[(wc * 64 + ct * 16 + l16) * LDA + g * 8]);
      acc[0][ct] = wmma16(a0, bf, acc[0][ct]);
      acc[1][ct] = wmma16(a1, bf, acc[1][ct]);
    }
    __syncthreads();
  }

  #pragma unroll
  for (int rt = 0; rt < 2; ++rt)
    #pragma unroll
    for (int ct = 0; ct < 4; ++ct)
      #pragma unroll
      for (int r = 0; r < 8; ++r) {
        int m = mb + wr * 32 + rt * 16 + r + 8 * g;
        int n = nb + wc * 64 + ct * 16 + l16;
        out[(size_t)m * Dc + n] = acc[rt][ct][r];
      }
}

} // anonymous namespace

extern "C" void kernel_launch(void* const* d_in, const int* in_sizes, int n_in,
                              void* d_out, int out_size, void* d_ws, size_t ws_size,
                              hipStream_t stream) {
  (void)in_sizes; (void)n_in; (void)out_size; (void)ws_size;
  const float* x     = (const float*)d_in[0];
  const int*   pmask = (const int*)d_in[1];
  const float* wqkv  = (const float*)d_in[2];
  const float* wout  = (const float*)d_in[3];
  float* out = (float*)d_out;

  const size_t headElems = (size_t)Bc * Hc * Sc * Dhc;   // 6,291,456
  _Float16* Qw = (_Float16*)d_ws;
  _Float16* Kw = Qw + headElems;
  _Float16* Vw = Kw + headElems;
  _Float16* Cw = Vw + headElems;                         // ctx: 8192 x 768 f16

  qkv_rope_kernel<<<dim3(Bc * Sc / 128, Nqkv / 128), 256, 0, stream>>>(x, wqkv, Qw, Kw, Vw);
  swa_kernel<<<dim3(Sc / 128, Hc, Bc), 256, 0, stream>>>(Qw, Kw, Vw, pmask, Cw);
  out_proj_kernel<<<dim3(Bc * Sc / 128, Dc / 128), 256, 0, stream>>>(Cw, wout, out);
}
